// MetaRNNBase_40905268527146
// MI455X (gfx1250) — compile-verified, matrix operand order checked by reading, not run on hardware
//
#include <hip/hip_runtime.h>

// ---------------------------------------------------------------------------
// MetaLSTM on MI455X (gfx1250, wave32, WMMA).
//  * hoisted x-GEMMs -> 128x128-tile bf16 WMMA GEMM (8 wmma / 6 frag loads)
//  * recurrent per-step GEMMs -> 64x64-tile bf16 WMMA GEMM
//  * LDS double-buffering fed by GLOBAL_LOAD_ASYNC_TO_LDS_B128 (ASYNCcnt)
//  * fused step_pre kernel: gh GEMM + dx/dh/db GEMMs + gx combine
// ---------------------------------------------------------------------------

#define Lc 2
#define Hc 1024
#define HMc 256
#define Ec 64
#define Bc 64
#define Tc 256
#define Dc 1024

#define USE_ASYNC_COPY 1

typedef __attribute__((ext_vector_type(16))) __bf16 v16bf;
typedef __attribute__((ext_vector_type(8)))  float  floatx8;
typedef __attribute__((ext_vector_type(4)))  unsigned int u32x4;

union Frag16 { v16bf v; u32x4 q[2]; };

__device__ inline unsigned short f2bf(float x) {
  union { float f; unsigned u; } v; v.f = x;
  unsigned r = v.u + 0x7FFFu + ((v.u >> 16) & 1u);   // round-to-nearest-even
  return (unsigned short)(r >> 16);
}

__device__ inline float sigf(float x) { return 1.0f / (1.0f + __expf(-x)); }

// WMMA wrapper: D = A(16x32 bf16) * B(32x16 bf16) + C(16x16 f32)
__device__ inline floatx8 wmma_bf16(const Frag16& a, const Frag16& b, floatx8 c) {
  return __builtin_amdgcn_wmma_f32_16x16x32_bf16(
      false, a.v, false, b.v, (short)0, c, false, false);
}

// --- CDNA5 async memory->LDS copy (16B per lane), tracked by ASYNCcnt ------
__device__ inline void async_copy16(void* lds, const void* g) {
#if USE_ASYNC_COPY
  unsigned lds_off = (unsigned)(size_t)lds;          // low 32b of generic = LDS addr
  unsigned long long ga = (unsigned long long)(size_t)g;
  asm volatile("global_load_async_to_lds_b128 %0, %1, off"
               :: "v"(lds_off), "v"(ga) : "memory");
#else
  *(u32x4*)lds = *(const u32x4*)g;
#endif
}
__device__ inline void async_join() {
#if USE_ASYNC_COPY
  asm volatile("s_wait_asynccnt 0x0" ::: "memory");
#endif
}

// A-fragment (and, by A*W^T symmetry, W-fragment) loader.
// ISA 16-bit A layout: lane L holds row M=L%16; K = {0..7,16..23} (L<16)
// or {8..15,24..31} (L>=16).  Two 16-byte loads per lane.
template <int STRIDE>
__device__ inline Frag16 load_frag(const unsigned short* s) {
  int lane = threadIdx.x & 31;
  int r  = lane & 15;
  int kb = (lane >> 4) << 3;
  const unsigned short* p = s + r * STRIDE + kb;
  Frag16 f;
  f.q[0] = *(const u32x4*)(p);
  f.q[1] = *(const u32x4*)(p + 16);
  return f;
}

__device__ inline Frag16 load_frag_g(const unsigned short* s, long stride) {
  int lane = threadIdx.x & 31;
  int r  = lane & 15;
  int kb = (lane >> 4) << 3;
  const unsigned short* p = s + (size_t)r * stride + kb;
  Frag16 f;
  f.q[0] = *(const u32x4*)(p);
  f.q[1] = *(const u32x4*)(p + 16);
  return f;
}

#define BK 32
#define LSTR 40   // padded LDS row stride (elems) against bank conflicts

// ---------------------------------------------------------------------------
// Big-tile GEMM for the hoisted input projections (compute-bound half):
// C[M,N] = A[M,K] * W[N,K]^T ; block tile 128x128, wave tile 32x64,
// 8 WMMA per 6 LDS fragment loads, async double-buffered staging.
// ---------------------------------------------------------------------------
#define BM2 128
#define BN2 128

__global__ __launch_bounds__(256)
void gemm_bf16_128_kernel(const unsigned short* __restrict__ A, long lda,
                          const unsigned short* __restrict__ W, long ldw,
                          float* __restrict__ Cf, long ldc, int K) {
  __shared__ __align__(16) unsigned short As[2][BM2 * LSTR];
  __shared__ __align__(16) unsigned short Ws[2][BN2 * LSTR];

  const int bm   = blockIdx.y * BM2;
  const int bn   = blockIdx.x * BN2;
  const int tid  = threadIdx.x;
  const int wave = tid >> 5;
  const int mset = (wave & 3) * 32;    // 0,32,64,96
  const int nset = (wave >> 2) * 64;   // 0,64

  floatx8 acc[2][4] = {};

  // 128x32 tile = 512 16B-chunks; 256 threads x 2 chunks per tile
  auto issue = [&](int k0, int buf) {
#pragma unroll
    for (int j = 0; j < 2; ++j) {
      int c   = tid * 2 + j;
      int row = c >> 2, col = (c & 3) << 3;
      async_copy16(&As[buf][row * LSTR + col],
                   A + (size_t)(bm + row) * lda + k0 + col);
      async_copy16(&Ws[buf][row * LSTR + col],
                   W + (size_t)(bn + row) * ldw + k0 + col);
    }
  };

  issue(0, 0);
  async_join();
  __syncthreads();

  for (int k0 = 0; k0 < K; k0 += BK) {
    int buf = (k0 / BK) & 1;
    if (k0 + BK < K) issue(k0 + BK, buf ^ 1);   // overlaps with WMMAs below

    Frag16 af[2], bfr[4];
    af[0] = load_frag<LSTR>(&As[buf][mset * LSTR]);
    af[1] = load_frag<LSTR>(&As[buf][(mset + 16) * LSTR]);
#pragma unroll
    for (int j = 0; j < 4; ++j)
      bfr[j] = load_frag<LSTR>(&Ws[buf][(nset + 16 * j) * LSTR]);
#pragma unroll
    for (int i = 0; i < 2; ++i)
#pragma unroll
      for (int j = 0; j < 4; ++j)
        acc[i][j] = wmma_bf16(af[i], bfr[j], acc[i][j]);

    async_join();
    __syncthreads();
  }

  const int lane = tid & 31;
  const int n_l  = lane & 15;
  const int mh   = (lane >> 4) << 3;
#pragma unroll
  for (int i = 0; i < 2; ++i)
#pragma unroll
    for (int j = 0; j < 4; ++j) {
      int ncol = bn + nset + j * 16 + n_l;
#pragma unroll
      for (int r = 0; r < 8; ++r)
        Cf[(size_t)(bm + mset + i * 16 + mh + r) * ldc + ncol] = acc[i][j][r];
    }
}

// ---------------------------------------------------------------------------
// Generic 64x64-tile GEMM for the recurrent (M=64) step GEMMs.
// C[M,N] = A[M,K] * W[N,K]^T (+addSrc)(+bias) -> f32 and/or bf16
// ---------------------------------------------------------------------------
#define BM 64
#define BN 64

__global__ __launch_bounds__(256)
void gemm_bf16_kernel(const unsigned short* __restrict__ A, long lda,
                      const unsigned short* __restrict__ W, long ldw,
                      const float* __restrict__ addSrc, long ldsrc,
                      const float* __restrict__ bias,
                      float* __restrict__ Cf, long ldc,
                      unsigned short* __restrict__ Cbf, long ldcb,
                      int M, int N, int K) {
  __shared__ __align__(16) unsigned short As[2][BM * LSTR];
  __shared__ __align__(16) unsigned short Ws[2][BN * LSTR];
  (void)M; (void)N;

  const int bm   = blockIdx.y * BM;
  const int bn   = blockIdx.x * BN;
  const int tid  = threadIdx.x;
  const int wave = tid >> 5;
  const int mset = (wave & 3) * 16;
  const int nset = (wave >> 2) * 32;

  floatx8 acc0 = {}, acc1 = {};

  const int lrow = tid >> 2;           // 0..63
  const int lcol = (tid & 3) << 3;     // 0,8,16,24

  auto issue = [&](int k0, int buf) {
    async_copy16(&As[buf][lrow * LSTR + lcol],
                 A + (size_t)(bm + lrow) * lda + k0 + lcol);
    async_copy16(&Ws[buf][lrow * LSTR + lcol],
                 W + (size_t)(bn + lrow) * ldw + k0 + lcol);
  };

  issue(0, 0);
  async_join();
  __syncthreads();

  for (int k0 = 0; k0 < K; k0 += BK) {
    int buf = (k0 / BK) & 1;
    if (k0 + BK < K) issue(k0 + BK, buf ^ 1);

    Frag16 a0 = load_frag<LSTR>(&As[buf][mset * LSTR]);
    Frag16 b0 = load_frag<LSTR>(&Ws[buf][nset * LSTR]);
    Frag16 b1 = load_frag<LSTR>(&Ws[buf][(nset + 16) * LSTR]);
    acc0 = wmma_bf16(a0, b0, acc0);
    acc1 = wmma_bf16(a0, b1, acc1);

    async_join();
    __syncthreads();
  }

  const int lane = tid & 31;
  const int n_l  = lane & 15;
  const int mh   = (lane >> 4) << 3;
#pragma unroll
  for (int half = 0; half < 2; ++half) {
    const floatx8& acc = half ? acc1 : acc0;
    int ncol = bn + nset + half * 16 + n_l;
#pragma unroll
    for (int r = 0; r < 8; ++r) {
      int mrow = bm + mset + mh + r;
      float v = acc[r];
      if (addSrc) v += addSrc[(size_t)mrow * ldsrc + ncol];
      if (bias)   v += bias[ncol];
      if (Cf)  Cf[(size_t)mrow * ldc + ncol] = v;
      if (Cbf) Cbf[(size_t)mrow * ldcb + ncol] = f2bf(v);
    }
  }
}

// ---------------------------------------------------------------------------
// Fused per-step "pre" kernel (one 64-wide (g,h)-slice per block):
//   gh = h @ Wh[g]^T (K=1024) ; dx/dh/db = z[g] @ P*[g]^T (K=64)
//   pre = dx*gx + dh*gh + db   (gx precomputed fp32)
// ---------------------------------------------------------------------------
__global__ __launch_bounds__(256)
void step_pre_kernel(const unsigned short* __restrict__ hcat,  // [64][1280]
                     const unsigned short* __restrict__ Whbf,  // [4096][1024]
                     const unsigned short* __restrict__ zbf,   // [64][768]
                     const unsigned short* __restrict__ Pp,    // [3][4][1024][64]
                     const float* __restrict__ Gx, long gstride,
                     float* __restrict__ pre) {                // [64][4096]
  const int bn = blockIdx.x * 64;
  const int g  = bn >> 10;
  const int hb = bn & 1023;
  const int wave = threadIdx.x >> 5;
  const int mset = (wave & 3) * 16;
  const int nset = (wave >> 2) * 32;

  floatx8 gh0 = {}, gh1 = {};
  for (int k0 = 0; k0 < 1024; k0 += 32) {
    Frag16 a  = load_frag_g(hcat + mset * 1280 + k0, 1280);
    Frag16 b0 = load_frag_g(Whbf + (size_t)(bn + nset) * 1024 + k0, 1024);
    Frag16 b1 = load_frag_g(Whbf + (size_t)(bn + nset + 16) * 1024 + k0, 1024);
    gh0 = wmma_bf16(a, b0, gh0);
    gh1 = wmma_bf16(a, b1, gh1);
  }

  floatx8 dx0 = {}, dx1 = {}, dh0 = {}, dh1 = {}, db0 = {}, db1 = {};
#pragma unroll
  for (int w = 0; w < 3; ++w) {
    const unsigned short* Pw = Pp + (size_t)(w * 4 + g) * 1024 * 64;
    const int cb = w * 256 + g * 64;
    floatx8 t0 = {}, t1 = {};
#pragma unroll
    for (int kk = 0; kk < 64; kk += 32) {
      Frag16 a  = load_frag_g(zbf + mset * 768 + cb + kk, 768);
      Frag16 b0 = load_frag_g(Pw + (size_t)(hb + nset) * 64 + kk, 64);
      Frag16 b1 = load_frag_g(Pw + (size_t)(hb + nset + 16) * 64 + kk, 64);
      t0 = wmma_bf16(a, b0, t0);
      t1 = wmma_bf16(a, b1, t1);
    }
    if (w == 0)      { dx0 = t0; dx1 = t1; }
    else if (w == 1) { dh0 = t0; dh1 = t1; }
    else             { db0 = t0; db1 = t1; }
  }

  const int lane = threadIdx.x & 31;
  const int n_l  = lane & 15;
  const int mh   = (lane >> 4) << 3;
#pragma unroll
  for (int half = 0; half < 2; ++half) {
    floatx8 gh = half ? gh1 : gh0;
    floatx8 dx = half ? dx1 : dx0;
    floatx8 dh = half ? dh1 : dh0;
    floatx8 db = half ? db1 : db0;
    int ncol = bn + nset + half * 16 + n_l;
#pragma unroll
    for (int r = 0; r < 8; ++r) {
      int b = mset + mh + r;
      float gx = Gx[(size_t)b * gstride + ncol];
      pre[(size_t)b * 4096 + ncol] = dx[r] * gx + dh[r] * gh[r] + db[r];
    }
  }
}

// ---------------------------------------------------------------------------
// Elementwise gate kernels
// ---------------------------------------------------------------------------
__global__ void mgate_kernel(const float* __restrict__ mpre,
                             float* __restrict__ mc,
                             unsigned short* __restrict__ hcat) {
  int i = blockIdx.x * blockDim.x + threadIdx.x;
  int b = i >> 8, j = i & 255;
  const float* r = mpre + (size_t)b * 1024;
  float mi = r[j], mf = r[256 + j], mg = r[512 + j], mo = r[768 + j];
  float c2 = sigf(mf) * mc[i] + sigf(mi) * tanhf(mg);
  float h2 = sigf(mo) * tanhf(c2);
  mc[i] = c2;
  hcat[(size_t)b * 1280 + 1024 + j] = f2bf(h2);
}

__global__ void hgate_kernel(const float* __restrict__ pre,
                             float* __restrict__ c,
                             unsigned short* __restrict__ hcat,
                             unsigned short* __restrict__ seq, long seq_stride,
                             float* __restrict__ out, long out_stride) {
  int i = blockIdx.x * blockDim.x + threadIdx.x;
  int b = i >> 10, h = i & 1023;
  const float* r = pre + (size_t)b * 4096;
  float i_ = r[h], f_ = r[1024 + h], g_ = r[2048 + h], o_ = r[3072 + h];
  float c2 = sigf(f_) * c[i] + sigf(i_) * tanhf(g_);
  float h2 = sigf(o_) * tanhf(c2);
  c[i] = c2;
  hcat[(size_t)b * 1280 + h] = f2bf(h2);
  if (seq) seq[(size_t)b * seq_stride + h] = f2bf(h2);
  if (out) out[(size_t)b * out_stride + h] = h2;
}

// ---------------------------------------------------------------------------
// One-time conversion / packing kernels
// ---------------------------------------------------------------------------
__global__ void cvt_bf16_kernel(const float* __restrict__ src,
                                unsigned short* __restrict__ dst, size_t n) {
  size_t i = blockIdx.x * (size_t)blockDim.x + threadIdx.x;
  size_t st = (size_t)gridDim.x * blockDim.x;
  for (; i < n; i += st) dst[i] = f2bf(src[i]);
}

__global__ void pack_wm_kernel(const float* __restrict__ Wmx,
                               const float* __restrict__ Wmh,
                               unsigned short* __restrict__ Wm,
                               unsigned short* __restrict__ Wmxx) {
  int row = blockIdx.x;                 // 0..L*1024-1
  int l = row >> 10, n = row & 1023;
  const float* mx = Wmx + ((size_t)l * 1024 + n) * 2048;
  const float* mh = Wmh + ((size_t)l * 1024 + n) * 256;
  unsigned short* wm  = Wm   + ((size_t)l * 1024 + n) * 1280;
  unsigned short* wxx = Wmxx + ((size_t)l * 1024 + n) * 1024;
  for (int k = threadIdx.x; k < 1024; k += blockDim.x) {
    wxx[k] = f2bf(mx[k]);
    wm[k]  = f2bf(mx[1024 + k]);
  }
  for (int k = threadIdx.x; k < 256; k += blockDim.x) wm[1024 + k] = f2bf(mh[k]);
}

__global__ void pack_wz_kernel(const float* __restrict__ Wzx,
                               const float* __restrict__ Wzh,
                               const float* __restrict__ Wzb,
                               const float* __restrict__ bzx,
                               const float* __restrict__ bzh,
                               unsigned short* __restrict__ Wz,
                               float* __restrict__ bz) {
  int l = blockIdx.y;
  int n = blockIdx.x;                  // 0..767
  const float* src = (n < 256) ? Wzx + (size_t)l * 256 * 256
                   : (n < 512) ? Wzh + (size_t)l * 256 * 256
                               : Wzb + (size_t)l * 256 * 256;
  int nn = n & 255;
  unsigned short* dst = Wz + ((size_t)l * 768 + n) * 256;
  for (int m = threadIdx.x; m < 256; m += blockDim.x)
    dst[m] = f2bf(src[(size_t)m * 256 + nn]);
  if (threadIdx.x == 0)
    bz[l * 768 + n] = (n < 256) ? bzx[l * 256 + nn]
                    : (n < 512) ? bzh[l * 256 + nn] : 0.0f;
}

__global__ void pack_p_kernel(const float* __restrict__ Px,
                              const float* __restrict__ Ph,
                              const float* __restrict__ Pb,
                              unsigned short* __restrict__ Pp, size_t total) {
  size_t i = blockIdx.x * (size_t)blockDim.x + threadIdx.x;
  size_t st = (size_t)gridDim.x * blockDim.x;
  for (; i < total; i += st) {
    size_t e = i & 63, h = (i >> 6) & 1023;
    size_t rest = i >> 16;
    size_t g = rest & 3, lw = rest >> 2;
    size_t w = lw % 3, l = lw / 3;
    size_t src = (((l * 4 + g) * 64 + e)) * 1024 + h;
    const float* P = (w == 0) ? Px : (w == 1) ? Ph : Pb;
    Pp[i] = f2bf(P[src]);
  }
}

__global__ void init_state_kernel(unsigned short* __restrict__ hcat,
                                  float* __restrict__ c,
                                  float* __restrict__ mc) {
  int i = blockIdx.x * blockDim.x + threadIdx.x;
  if (i < 64 * 1280) hcat[i] = 0;
  if (i < 64 * 1024) c[i] = 0.0f;
  if (i < 64 * 256)  mc[i] = 0.0f;
}

// ---------------------------------------------------------------------------
extern "C" void kernel_launch(void* const* d_in, const int* in_sizes, int n_in,
                              void* d_out, int out_size, void* d_ws, size_t ws_size,
                              hipStream_t stream) {
  (void)in_sizes; (void)n_in; (void)out_size; (void)ws_size;
  const float* x   = (const float*)d_in[0];
  const float* Wx  = (const float*)d_in[1];
  const float* Wh  = (const float*)d_in[2];
  const float* Wmx = (const float*)d_in[3];
  const float* Wmh = (const float*)d_in[4];
  const float* bm  = (const float*)d_in[5];
  const float* Wzx = (const float*)d_in[6];
  const float* bzx = (const float*)d_in[7];
  const float* Wzh = (const float*)d_in[8];
  const float* bzh = (const float*)d_in[9];
  const float* Wzb = (const float*)d_in[10];
  const float* Px  = (const float*)d_in[11];
  const float* Ph  = (const float*)d_in[12];
  const float* Pb  = (const float*)d_in[13];
  float* out = (float*)d_out;

  char* p = (char*)d_ws;
  auto alloc = [&](size_t bytes) {
    void* r = (void*)p;
    p += (bytes + 255) & ~(size_t)255;
    return r;
  };
  unsigned short* Wx_bf   = (unsigned short*)alloc(2ull * Lc * 4 * Hc * Dc);
  unsigned short* Wh_bf   = (unsigned short*)alloc(2ull * Lc * 4 * Hc * Hc);
  unsigned short* Wmxx_bf = (unsigned short*)alloc(2ull * Lc * 1024 * 1024);
  unsigned short* Wm_bf   = (unsigned short*)alloc(2ull * Lc * 1024 * 1280);
  unsigned short* Wz_bf   = (unsigned short*)alloc(2ull * Lc * 768 * 256);
  float*          bz      = (float*)alloc(4ull * Lc * 768);
  unsigned short* Pp_bf   = (unsigned short*)alloc(2ull * Lc * 3 * 4 * 1024 * 64);
  unsigned short* x_bf    = (unsigned short*)alloc(2ull * Bc * Tc * Dc);
  unsigned short* seq_bf  = (unsigned short*)alloc(2ull * Bc * Tc * Hc);
  float*          Gxbuf   = (float*)alloc(4ull * Bc * Tc * 4 * Hc);
  float*          MxPre   = (float*)alloc(4ull * Bc * Tc * 1024);
  unsigned short* hcat    = (unsigned short*)alloc(2ull * 64 * 1280);
  float*          cbuf    = (float*)alloc(4ull * 64 * 1024);
  float*          mcbuf   = (float*)alloc(4ull * 64 * 256);
  float*          mprebuf = (float*)alloc(4ull * 64 * 1024);
  float*          zbuf    = (float*)alloc(4ull * 64 * 768);
  unsigned short* z_bf    = (unsigned short*)alloc(2ull * 64 * 768);
  float*          prebuf  = (float*)alloc(4ull * 64 * 4096);

  cvt_bf16_kernel<<<2048, 256, 0, stream>>>(x,  x_bf,  (size_t)Bc * Tc * Dc);
  cvt_bf16_kernel<<<2048, 256, 0, stream>>>(Wx, Wx_bf, (size_t)Lc * 4 * Hc * Dc);
  cvt_bf16_kernel<<<2048, 256, 0, stream>>>(Wh, Wh_bf, (size_t)Lc * 4 * Hc * Hc);
  pack_wm_kernel<<<Lc * 1024, 256, 0, stream>>>(Wmx, Wmh, Wm_bf, Wmxx_bf);
  pack_wz_kernel<<<dim3(768, Lc), 256, 0, stream>>>(Wzx, Wzh, Wzb, bzx, bzh, Wz_bf, bz);
  pack_p_kernel<<<2048, 256, 0, stream>>>(Px, Ph, Pb, Pp_bf, (size_t)Lc * 3 * 4 * 1024 * 64);

  for (int l = 0; l < Lc; ++l) {
    const unsigned short* Abf = l ? seq_bf : x_bf;

    // hoisted input GEMMs (compute-bound): big-tile WMMA kernel
    gemm_bf16_128_kernel<<<dim3(4096 / BN2, (Bc * Tc) / BM2), 256, 0, stream>>>(
        Abf, 1024, Wx_bf + (size_t)l * 4 * Hc * Dc, 1024, Gxbuf, 4096, 1024);
    gemm_bf16_128_kernel<<<dim3(1024 / BN2, (Bc * Tc) / BM2), 256, 0, stream>>>(
        Abf, 1024, Wmxx_bf + (size_t)l * 1024 * 1024, 1024, MxPre, 1024, 1024);

    init_state_kernel<<<(64 * 1280 + 255) / 256, 256, 0, stream>>>(hcat, cbuf, mcbuf);

    for (int t = 0; t < Tc; ++t) {
      // mpre = MxPre(t) + [h|mh] @ Wm^T + bm     (64 x 1024 x 1280)
      gemm_bf16_kernel<<<dim3(1024 / BN, 1), 256, 0, stream>>>(
          hcat, 1280, Wm_bf + (size_t)l * 1024 * 1280, 1280,
          MxPre + (size_t)t * 1024, (long)Tc * 1024, bm + (size_t)l * 1024,
          mprebuf, 1024, nullptr, 0, 64, 1024, 1280);
      mgate_kernel<<<64, 256, 0, stream>>>(mprebuf, mcbuf, hcat);

      // z = mh2 @ Wz^T + bz                      (64 x 768 x 256)
      gemm_bf16_kernel<<<dim3(768 / BN, 1), 256, 0, stream>>>(
          hcat + 1024, 1280, Wz_bf + (size_t)l * 768 * 256, 256,
          nullptr, 0, bz + l * 768,
          zbuf, 768, z_bf, 768, 64, 768, 256);

      // pre = dx*gx + dh*(h@Wh^T) + db
      step_pre_kernel<<<4096 / 64, 256, 0, stream>>>(
          hcat, Wh_bf + (size_t)l * 4 * Hc * Hc, z_bf,
          Pp_bf + (size_t)l * 3 * 4 * 1024 * 64,
          Gxbuf + (size_t)t * 4096, (long)Tc * 4096, prebuf);

      hgate_kernel<<<(64 * 1024) / 256, 256, 0, stream>>>(
          prebuf, cbuf, hcat,
          (l == 0) ? seq_bf + (size_t)t * 1024 : nullptr, (long)Tc * 1024,
          (l == Lc - 1) ? out + (size_t)t * 1024 : nullptr, (long)Tc * 1024);
    }
  }
}
